// SelfAttention_14181982012236
// MI455X (gfx1250) — matrix operand order repaired and checked by reading, not compiled, not run on hardware
//
#include <hip/hip_runtime.h>
#include <hip/hip_bf16.h>

// ---------------------------------------------------------------------------
// Self-attention (B=1, T=2048, 32 heads, d=128, GQA 8 groups, RoPE first 64)
// Pipeline:
//   1) f32->f16 convert of x, Wqkv, Wout
//   2) QKV GEMM (WMMA f16, f32 accum, async-LDS double-buffered) -> qkv f16
//   3) RoPE + GQA rearrange -> Qh[32][2048][128], Kh[8][2048][128], Vt[8][128][2048]
//   4) Flash attention, scores computed transposed (St = K*Q^T) so P maps to
//      the PV WMMA A-operand layout in-lane (zero cross-lane traffic).
//   5) Output GEMM (WMMA) -> d_out f32 [2048,4096]
// ---------------------------------------------------------------------------

typedef __attribute__((ext_vector_type(16))) _Float16 v16h;
typedef __attribute__((ext_vector_type(8)))  _Float16 v8h;
typedef __attribute__((ext_vector_type(4)))  _Float16 v4h;
typedef __attribute__((ext_vector_type(8)))  float    v8f;
typedef __attribute__((ext_vector_type(4)))  float    v4f;
typedef __attribute__((ext_vector_type(4)))  int      v4i;

union V16H { v16h v; v8h h[2]; };

#if defined(__gfx1250__) && __has_builtin(__builtin_amdgcn_global_load_async_to_lds_b128)
#define HAVE_ASYNC_LDS 1
#else
#define HAVE_ASYNC_LDS 0
#endif

__device__ __forceinline__ v8f wmma_f16(v16h a, v16h b, v8f c) {
    // (neg_a, A, neg_b, B, c_mod, C, reuse_a, reuse_b)
    return __builtin_amdgcn_wmma_f32_16x16x32_f16(false, a, false, b, (short)0, c,
                                                  false, false);
}
__device__ __forceinline__ v8f zero8() {
    v8f z;
#pragma unroll
    for (int i = 0; i < 8; ++i) z[i] = 0.0f;
    return z;
}

__device__ __forceinline__ void wait_async0() {
#if HAVE_ASYNC_LDS
#if __has_builtin(__builtin_amdgcn_s_wait_asynccnt)
    __builtin_amdgcn_s_wait_asynccnt(0);
#else
    asm volatile("s_wait_asynccnt 0" ::: "memory");
#endif
#endif
}

#if HAVE_ASYNC_LDS
__device__ __forceinline__ void async_b128(const _Float16* gsrc, _Float16* ldst) {
    // prototype: (global v4i* src, local v4i* dst, imm offset, imm cpol)
    __builtin_amdgcn_global_load_async_to_lds_b128(
        (__attribute__((address_space(1))) v4i*)gsrc,
        (__attribute__((address_space(3))) v4i*)ldst, 0, 0);
}
#endif

// ---------------------------------------------------------------------------
// 1) f32 -> f16 conversion (vectorized; n is a multiple of 4)
// ---------------------------------------------------------------------------
__global__ void cvt_f32_to_f16_kernel(const float* __restrict__ in,
                                      _Float16* __restrict__ out, long n) {
    long i = ((long)blockIdx.x * blockDim.x + threadIdx.x) * 4;
    if (i < n) {
        v4f x = *(const v4f*)(in + i);
        v4h y;
#pragma unroll
        for (int j = 0; j < 4; ++j) y[j] = (_Float16)x[j];
        *(v4h*)(out + i) = y;
    }
}

// ---------------------------------------------------------------------------
// 2/5) Generic f16 GEMM: C[M,N] = A[M,K] * B[N,K]^T  (A,B row-major f16)
//      Block tile 128x128, 8 waves (wave32), each wave 32x64 (2x4 WMMA tiles).
//      K staged 32-wide through double-buffered LDS; fills via async-to-LDS
//      (ASYNCcnt) when available, else global->VGPR->LDS with reg prefetch.
//      One barrier per K step: a wave writes buffer `nxt` only after barrier i,
//      and all waves' reads of that buffer completed before reaching barrier i.
// ---------------------------------------------------------------------------
#define LDSTR 40   // halves per LDS row (32 data + 8 pad): 80B, 16B aligned, conflict-free

template <bool F16OUT>
__global__ __launch_bounds__(256) void gemm_f16_kernel(
    const _Float16* __restrict__ A, const _Float16* __restrict__ B,
    void* __restrict__ Cout, int M, int N, int K) {
    __shared__ __align__(16) _Float16 As[2][128 * LDSTR];
    __shared__ __align__(16) _Float16 Bs[2][128 * LDSTR];

    const int tid  = threadIdx.x;
    const int lane = tid & 31;
    const int wave = tid >> 5;
    const int l16  = lane & 15;
    const int hi   = lane >> 4;          // half-wave select
    const int m0   = blockIdx.y * 128;
    const int n0   = blockIdx.x * 128;
    const int wr   = wave >> 1;          // 0..3 -> wave row (32 rows each)
    const int wc   = wave & 1;           // 0..1 -> wave col (64 cols each)

    v8f acc[2][4];
#pragma unroll
    for (int mt = 0; mt < 2; ++mt)
#pragma unroll
        for (int nt = 0; nt < 4; ++nt) acc[mt][nt] = zero8();

    // Each thread stages 32B of A and 32B of B per K step.
    const int srow = tid >> 1;           // 0..127
    const int scol = (tid & 1) * 16;     // 0 or 16 (halves)
    const _Float16* agBase = A + (size_t)(m0 + srow) * K + scol;
    const _Float16* bgBase = B + (size_t)(n0 + srow) * K + scol;
    const int sidx = srow * LDSTR + scol;

    auto compute_tile = [&](const _Float16* as, const _Float16* bs) {
        V16H af[2], bf[4];
#pragma unroll
        for (int mt = 0; mt < 2; ++mt) {
            int m = wr * 32 + mt * 16 + l16;
            af[mt].h[0] = *(const v8h*)&as[m * LDSTR + hi * 8];       // K {0..7}/{8..15}
            af[mt].h[1] = *(const v8h*)&as[m * LDSTR + 16 + hi * 8];  // K {16..23}/{24..31}
        }
#pragma unroll
        for (int nt = 0; nt < 4; ++nt) {
            int n = wc * 64 + nt * 16 + l16;
            bf[nt].h[0] = *(const v8h*)&bs[n * LDSTR + hi * 8];       // K {0..7}/{8..15}
            bf[nt].h[1] = *(const v8h*)&bs[n * LDSTR + 16 + hi * 8];  // K {16..23}/{24..31}
        }
#pragma unroll
        for (int mt = 0; mt < 2; ++mt)
#pragma unroll
            for (int nt = 0; nt < 4; ++nt)
                acc[mt][nt] = wmma_f16(af[mt].v, bf[nt].v, acc[mt][nt]);
    };

#if HAVE_ASYNC_LDS
    // ---- async-to-LDS double-buffered pipeline --------------------------
    auto stage_async = [&](int k0, int buf) {
        const _Float16* ag = agBase + k0;
        const _Float16* bg = bgBase + k0;
        async_b128(ag,     &As[buf][sidx]);
        async_b128(ag + 8, &As[buf][sidx + 8]);
        async_b128(bg,     &Bs[buf][sidx]);
        async_b128(bg + 8, &Bs[buf][sidx + 8]);
    };
    stage_async(0, 0);
    for (int k0 = 0; k0 < K; k0 += 32) {
        const int cur = (k0 >> 5) & 1;
        wait_async0();          // this wave's fills of buf[cur] have landed
        __syncthreads();        // everyone's fills landed; prev compute done
        if (k0 + 32 < K) stage_async(k0 + 32, cur ^ 1);  // overlap with WMMA
        compute_tile(As[cur], Bs[cur]);
    }
#else
    // ---- synchronous fallback: global->VGPR prefetch, single barrier ----
    v8h ra0 = *(const v8h*)(agBase);
    v8h ra1 = *(const v8h*)(agBase + 8);
    v8h rb0 = *(const v8h*)(bgBase);
    v8h rb1 = *(const v8h*)(bgBase + 8);
    for (int k0 = 0; k0 < K; k0 += 32) {
        const int cur = (k0 >> 5) & 1;
        *(v8h*)&As[cur][sidx]     = ra0;
        *(v8h*)&As[cur][sidx + 8] = ra1;
        *(v8h*)&Bs[cur][sidx]     = rb0;
        *(v8h*)&Bs[cur][sidx + 8] = rb1;
        if (k0 + 32 < K) {      // fetch next tile while barrier+compute run
            const _Float16* ag = agBase + k0 + 32;
            const _Float16* bg = bgBase + k0 + 32;
            ra0 = *(const v8h*)(ag);
            ra1 = *(const v8h*)(ag + 8);
            rb0 = *(const v8h*)(bg);
            rb1 = *(const v8h*)(bg + 8);
        }
        __syncthreads();
        compute_tile(As[cur], Bs[cur]);
    }
#endif

    // C layout: lane holds column l16; VGPR r holds row r + 8*hi.
#pragma unroll
    for (int mt = 0; mt < 2; ++mt)
#pragma unroll
        for (int nt = 0; nt < 4; ++nt) {
            int col = n0 + wc * 64 + nt * 16 + l16;
#pragma unroll
            for (int r = 0; r < 8; ++r) {
                int row = m0 + wr * 32 + mt * 16 + r + 8 * hi;
                if (F16OUT)
                    ((_Float16*)Cout)[(size_t)row * N + col] = (_Float16)acc[mt][nt][r];
                else
                    ((float*)Cout)[(size_t)row * N + col] = acc[mt][nt][r];
            }
        }
}

// ---------------------------------------------------------------------------
// 3) RoPE + GQA rearrange. qkv viewed as [t][g(8)][slot(6)][128];
//    slots 0-3 -> Q head g*4+slot, 4 -> K, 5 -> V (stored transposed).
// ---------------------------------------------------------------------------
__global__ void rope_rearrange_kernel(const _Float16* __restrict__ qkv,
                                      const float* __restrict__ cosb,
                                      const float* __restrict__ sinb,
                                      _Float16* __restrict__ Qh,
                                      _Float16* __restrict__ Kh,
                                      _Float16* __restrict__ Vt) {
    const int d    = threadIdx.x;        // 0..127
    const int idx  = blockIdx.x;         // t*48 + g*6 + slot
    const int slot = idx % 6;
    const int g    = (idx / 6) % 8;
    const int t    = idx / 48;

    const _Float16* row = qkv + (size_t)t * 6144 + (size_t)(g * 6 + slot) * 128;
    float v = (float)row[d];
    if (slot < 5 && d < 64) {            // RoPE on q & k, first 64 dims
        float pair = (float)row[d < 32 ? d + 32 : d - 32];
        float rot  = (d < 32) ? -pair : pair;
        v = v * cosb[t * 64 + d] + rot * sinb[t * 64 + d];
    }
    _Float16 hv = (_Float16)v;
    if (slot < 4)
        Qh[((size_t)(g * 4 + slot) * 2048 + t) * 128 + d] = hv;
    else if (slot == 4)
        Kh[((size_t)g * 2048 + t) * 128 + d] = hv;
    else
        Vt[((size_t)g * 128 + d) * 2048 + t] = hv;   // transposed: [d][t]
}

// ---------------------------------------------------------------------------
// 4) Flash attention. One wave handles 16 queries; block = 8 waves = 128
//    queries per head. Scores computed transposed: St = K_tile * Q^T so the
//    softmaxed P is already in WMMA A-operand layout (no LDS transpose).
// ---------------------------------------------------------------------------
__global__ __launch_bounds__(256) void attn_kernel(
    const _Float16* __restrict__ Qh, const _Float16* __restrict__ Kh,
    const _Float16* __restrict__ Vt, _Float16* __restrict__ Yh) {
    const int lane = threadIdx.x & 31;
    const int wave = threadIdx.x >> 5;
    const int l16  = lane & 15;
    const int hi   = lane >> 4;
    const int head = blockIdx.y;
    const int grp  = head >> 2;          // 4 query heads per KV group
    const int q0   = blockIdx.x * 128 + wave * 16;

    const _Float16* Qb = Qh + (size_t)head * 2048 * 128;
    const _Float16* Kb = Kh + (size_t)grp * 2048 * 128;
    const _Float16* Vb = Vt + (size_t)grp * 128 * 2048;

    // Q as B-operand fragments (lane = query column, contiguous 16 K-halves)
    V16H qf[4];
#pragma unroll
    for (int kc = 0; kc < 4; ++kc) {
        const _Float16* p = Qb + (size_t)(q0 + l16) * 128 + kc * 32 + hi * 16;
        qf[kc].h[0] = *(const v8h*)p;
        qf[kc].h[1] = *(const v8h*)(p + 8);
    }

    v8f acc[8];
#pragma unroll
    for (int dt = 0; dt < 8; ++dt) acc[dt] = zero8();
    float m_run = -1e30f, l_run = 0.0f;
    const float scale = 0.08838834764831845f;   // 1/sqrt(128)
    const int   q     = q0 + l16;               // this lane's query
    const int   kend  = q0 + 16;                // causal bound for the tile

    for (int k0 = 0; k0 < kend; k0 += 32) {
        // St tiles (16 keys x 16 queries each): A = K rows, B = Q^T
        v8f st[2] = {zero8(), zero8()};
#pragma unroll
        for (int t = 0; t < 2; ++t)
#pragma unroll
            for (int kc = 0; kc < 4; ++kc) {
                V16H kf;
                const _Float16* p =
                    Kb + (size_t)(k0 + t * 16 + l16) * 128 + kc * 32 + hi * 16;
                kf.h[0] = *(const v8h*)p;          // d {0..7}/{16..23}
                kf.h[1] = *(const v8h*)(p + 8);    // d {8..15}/{24..31}
                st[t]   = wmma_f16(kf.v, qf[kc].v, st[t]);
            }

        // scale + causal mask; element i <-> key k0 + t*16 + r + 8*hi
        float vals[16];
#pragma unroll
        for (int t = 0; t < 2; ++t)
#pragma unroll
            for (int r = 0; r < 8; ++r) {
                int key = k0 + t * 16 + r + 8 * hi;
                float s = st[t][r] * scale;
                vals[t * 8 + r] = (key <= q) ? s : -1e30f;
            }

        // online softmax: row = this lane's query (partner lane l^16 has the
        // other 16 keys of the block)
        float rm = vals[0];
#pragma unroll
        for (int i = 1; i < 16; ++i) rm = fmaxf(rm, vals[i]);
        rm = fmaxf(rm, __shfl_xor(rm, 16, 32));
        float m_new = fmaxf(m_run, rm);
        float corr  = __expf(m_run - m_new);

        V16H pf;                                   // P in A-operand layout
        float psum = 0.0f;
#pragma unroll
        for (int i = 0; i < 16; ++i) {
            float p = __expf(vals[i] - m_new);
            psum += p;
            pf.v[i] = (_Float16)p;
        }
        psum += __shfl_xor(psum, 16, 32);
        l_run = l_run * corr + psum;
        m_run = m_new;

        // broadcast corr to acc row owners (acc row = query r + 8*hi)
        float corr_b[8];
#pragma unroll
        for (int r = 0; r < 8; ++r) corr_b[r] = __shfl(corr, r + 8 * hi, 32);

        // P*V: B = V (lane = d column, keys contiguous from Vt[d][t])
#pragma unroll
        for (int dt = 0; dt < 8; ++dt) {
            V16H vf;
            const _Float16* p = Vb + (size_t)(dt * 16 + l16) * 2048 + k0 + hi * 8;
            vf.h[0] = *(const v8h*)p;
            vf.h[1] = *(const v8h*)(p + 16);
#pragma unroll
            for (int r = 0; r < 8; ++r) acc[dt][r] *= corr_b[r];
            acc[dt] = wmma_f16(pf.v, vf.v, acc[dt]);
        }
    }

    float linv_b[8];
#pragma unroll
    for (int r = 0; r < 8; ++r) linv_b[r] = 1.0f / __shfl(l_run, r + 8 * hi, 32);

#pragma unroll
    for (int dt = 0; dt < 8; ++dt)
#pragma unroll
        for (int r = 0; r < 8; ++r) {
            int qrow = q0 + r + 8 * hi;
            Yh[(size_t)qrow * 4096 + head * 128 + dt * 16 + l16] =
                (_Float16)(acc[dt][r] * linv_b[r]);
        }
}

// ---------------------------------------------------------------------------
// launcher
// ---------------------------------------------------------------------------
extern "C" void kernel_launch(void* const* d_in, const int* in_sizes, int n_in,
                              void* d_out, int out_size, void* d_ws, size_t ws_size,
                              hipStream_t stream) {
    const float* x    = (const float*)d_in[0];
    const float* Wqkv = (const float*)d_in[1];
    const float* Wout = (const float*)d_in[2];
    const float* cosb = (const float*)d_in[3];
    const float* sinb = (const float*)d_in[4];
    // d_in[5] (mask) unused: causal mask computed analytically.

    char* ws = (char*)d_ws;
    // buf1: Xh (before gemm1) then Qh (after rope) -- both 16 MB
    _Float16* Xh    = (_Float16*)ws;                 ws += (size_t)2048 * 4096 * 2;
    // buf2: QKVh (gemm1 out) then Yh (attn out, first 16 MB) -- 24 MB
    _Float16* QKVh  = (_Float16*)ws;                 ws += (size_t)2048 * 6144 * 2;
    _Float16* Wqkvh = (_Float16*)ws;                 ws += (size_t)6144 * 4096 * 2;
    _Float16* Wouth = (_Float16*)ws;                 ws += (size_t)4096 * 4096 * 2;
    _Float16* Kh    = (_Float16*)ws;                 ws += (size_t)8 * 2048 * 128 * 2;
    _Float16* Vt    = (_Float16*)ws;                 ws += (size_t)8 * 128 * 2048 * 2;
    _Float16* Qh    = Xh;      // alias: Xh dead after QKV GEMM
    _Float16* Yh    = QKVh;    // alias: QKVh dead after rearrange

    // 1) convert inputs to f16
    cvt_f32_to_f16_kernel<<<(2048L * 4096 / 4 + 255) / 256, 256, 0, stream>>>(
        x, Xh, 2048L * 4096);
    cvt_f32_to_f16_kernel<<<(6144L * 4096 / 4 + 255) / 256, 256, 0, stream>>>(
        Wqkv, Wqkvh, 6144L * 4096);
    cvt_f32_to_f16_kernel<<<(4096L * 4096 / 4 + 255) / 256, 256, 0, stream>>>(
        Wout, Wouth, 4096L * 4096);

    // 2) QKV projection
    gemm_f16_kernel<true><<<dim3(6144 / 128, 2048 / 128), 256, 0, stream>>>(
        Xh, Wqkvh, (void*)QKVh, 2048, 6144, 4096);

    // 3) RoPE + rearrange
    rope_rearrange_kernel<<<2048 * 48, 128, 0, stream>>>(QKVh, cosb, sinb, Qh, Kh, Vt);

    // 4) flash attention
    attn_kernel<<<dim3(2048 / 128, 32), 256, 0, stream>>>(Qh, Kh, Vt, Yh);

    // 5) output projection -> f32 d_out
    gemm_f16_kernel<false><<<dim3(4096 / 128, 2048 / 128), 256, 0, stream>>>(
        Yh, Wouth, d_out, 2048, 4096, 4096);
}
